// FSASFourierModule_20100446945426
// MI455X (gfx1250) — compile-verified, matrix operand order checked by reading, not run on hardware
//
#include <hip/hip_runtime.h>
#include <hip/hip_bf16.h>
#include <math.h>

// ---------------------------------------------------------------------------
// CDNA5 / gfx1250 implementation.  wave32; matrix ops via v_wmma_f32_16x16x32_f16.
// LDS tiles are stored in WMMA *fragment order* so each lane loads its v16h
// fragment as one aligned 32-byte read (2x ds_load_b128).  Block tile is
// M=32 x N=256: each wave holds 2 A fragments + 2 N-subtiles -> 4 wmma/K-step
// with B fragments reused across both M-subtiles.  Conv weights are converted
// to f16 once (prep kernels) so the GEMM hot loop has no f32->f16 converts.
// ---------------------------------------------------------------------------

typedef _Float16 f16;
typedef __attribute__((ext_vector_type(4)))  _Float16 f16x4;
typedef __attribute__((ext_vector_type(16))) _Float16 v16h;
typedef __attribute__((ext_vector_type(8)))  float    v8f;

static const int B_  = 4;
static const int C_  = 64;
static const int H_  = 256;
static const int W_  = 256;
static const int HW_ = H_ * W_;

__device__ __forceinline__ float gelu_exact(float v) {
  return 0.5f * v * (1.0f + erff(v * 0.70710678118654752f));
}
__device__ __forceinline__ float sigmoidf_(float v) {
  return 1.0f / (1.0f + __expf(-v));
}

// ISA 7.12.2 16-bit fragment mapping (wave32):
//   lane half 0: e=0..7 -> K=e,    e=8..15 -> K=e+8   (K 0..7, 16..23)
//   lane half 1: e=0..7 -> K=e+8,  e=8..15 -> K=e+16  (K 8..15, 24..31)
__device__ __forceinline__ void k_to_he(int k, int& half, int& e) {
  half = (k >> 3) & 1;
  e    = (k & 7) + ((k >> 4) << 3);
}

// ---------------------------------------------------------------------------
// Weight prep: fp32 -> f16 (one time).
// ---------------------------------------------------------------------------
__global__ void prep_w_f16(const float* __restrict__ src, f16* __restrict__ dst, int n) {
  int i = blockIdx.x * 256 + threadIdx.x;
  if (i < n) dst[i] = (f16)src[i];
}
// fus_conv1_w [128][128][9] -> [9][128][128] (tap-major, contiguous in ic)
__global__ void prep_w3x3(const float* __restrict__ src, f16* __restrict__ dst) {
  int i = blockIdx.x * 256 + threadIdx.x;         // over 147456
  if (i < 147456) {
    int tap = i % 9, oc_c = i / 9;
    dst[tap * 16384 + oc_c] = (f16)src[i];
  }
}

// ---------------------------------------------------------------------------
// 1x1 conv as WMMA GEMM: out[b][oc][p] = act(bn(W @ in))
// in : f16 [B][Cin][HW], w16 : f16 [Cout][Cin], out : f16 [B][Cout][HW]
// block = 256 threads (8 waves).  Block tile: M=32, N=256.
// grid = (HW/256, Cout/32, B)
// ---------------------------------------------------------------------------
__global__ void conv1x1_wmma(const f16* __restrict__ in, const f16* __restrict__ w16,
                             f16* __restrict__ out,
                             const float* __restrict__ scale, const float* __restrict__ shift,
                             float bn_mul, int Cin, int Cout, int act) {
  const int tid  = threadIdx.x;
  const int wave = tid >> 5;
  const int lane = tid & 31;
  const int half = lane >> 4;
  const int row  = lane & 15;
  const int b    = blockIdx.z;
  const int oc0  = blockIdx.y << 5;
  const long p0  = (long)blockIdx.x * 256;

  __shared__ __align__(32) f16 Af[2][32 * 16];     // 2 KB (two M-subtiles)
  __shared__ __align__(32) f16 Bf[16 * 32 * 16];   // 16 KB

  v8f zero = {0.f, 0.f, 0.f, 0.f, 0.f, 0.f, 0.f, 0.f};
  v8f acc[2][2] = {{zero, zero}, {zero, zero}};

  for (int k0 = 0; k0 < Cin; k0 += 32) {
    // ---- A tile: 32(M) x 32(K) f16 weights, one f16x4 group per thread
    {
      int m  = tid >> 3;             // 0..31
      int kg = (tid & 7) << 2;       // 0,4,...,28
      f16x4 wv = *(const f16x4*)&w16[(size_t)(oc0 + m) * Cin + k0 + kg];
      int mt = m >> 4, mm = m & 15;
#pragma unroll
      for (int u = 0; u < 4; ++u) {
        int h, e; k_to_he(kg + u, h, e);
        Af[mt][(((h << 4) | mm) << 4) + e] = wv[u];
      }
    }
    // ---- B tile: 32(K) x 256(N).  Load phase (pipelined) then scatter phase.
    f16x4 vals[8];
#pragma unroll
    for (int it = 0; it < 8; ++it) {
      int i = tid + (it << 8);
      int r = i >> 6, g = i & 63;
      vals[it] = *(const f16x4*)&in[((size_t)b * Cin + k0 + r) * HW_ + p0 + (g << 2)];
    }
#pragma unroll
    for (int it = 0; it < 8; ++it) {
      int i = tid + (it << 8);
      int r = i >> 6, g = i & 63;
      int h, e; k_to_he(r, h, e);
#pragma unroll
      for (int u = 0; u < 4; ++u) {
        int n = (g << 2) + u;
        int l = (h << 4) | (n & 15);
        Bf[((((n >> 4) << 5) + l) << 4) + e] = vals[it][u];
      }
    }
    if (k0 + 32 < Cin)  // pull next K-slab toward L2 (global_prefetch_b8)
      __builtin_prefetch(&in[((size_t)b * Cin + k0 + 32) * HW_ + p0], 0, 1);
    __syncthreads();

    v16h a0 = *(const v16h*)(Af[0] + (lane << 4));  // 2x ds_load_b128 each
    v16h a1 = *(const v16h*)(Af[1] + (lane << 4));
#pragma unroll
    for (int t2 = 0; t2 < 2; ++t2) {
      int nt = (wave << 1) + t2;
      v16h bf = *(const v16h*)(Bf + (((nt << 5) + lane) << 4));
      acc[0][t2] = __builtin_amdgcn_wmma_f32_16x16x32_f16(false, a0, false, bf,
                                                          (short)0, acc[0][t2], false, false);
      acc[1][t2] = __builtin_amdgcn_wmma_f32_16x16x32_f16(false, a1, false, bf,
                                                          (short)0, acc[1][t2], false, false);
    }
    __syncthreads();
  }

#pragma unroll
  for (int mt = 0; mt < 2; ++mt)
#pragma unroll
    for (int t2 = 0; t2 < 2; ++t2) {
      int nt = (wave << 1) + t2;
#pragma unroll
      for (int r = 0; r < 8; ++r) {
        int m   = (mt << 4) + r + (half << 3);
        float v = acc[mt][t2][r];
        if (scale) v *= scale[oc0 + m] * bn_mul;
        if (shift) v += shift[oc0 + m];
        if (act == 1) v = gelu_exact(v);
        out[((size_t)b * Cout + oc0 + m) * HW_ + p0 + (nt << 4) + row] = (f16)v;
      }
    }
}

// ---------------------------------------------------------------------------
// 3x3 conv (pad=1) as implicit GEMM, K = 9 taps x 128 ic.  Input is the
// virtual concat [inA(64ch) ; inB(64ch)].  Cout = 128, Cin = 128.
// w16 layout: [9][128 oc][128 ic].  Block tile: M=32 x N=256 (one image row).
// grid = (H, Cout/32 = 4, B), block 256.
// ---------------------------------------------------------------------------
__global__ void conv3x3_wmma(const f16* __restrict__ inA, const f16* __restrict__ inB,
                             const f16* __restrict__ w16, f16* __restrict__ out,
                             const float* __restrict__ scale, const float* __restrict__ shift,
                             float bn_mul, int act) {
  const int Cin = 128, Cout = 128;
  const int tid  = threadIdx.x;
  const int wave = tid >> 5;
  const int lane = tid & 31;
  const int half = lane >> 4;
  const int row  = lane & 15;
  const int b    = blockIdx.z;
  const int oc0  = blockIdx.y << 5;
  const int y    = blockIdx.x;            // output row
  const long p0  = (long)y * W_;

  __shared__ __align__(32) f16 Af[2][32 * 16];
  __shared__ __align__(32) f16 Bf[16 * 32 * 16];

  v8f zero = {0.f, 0.f, 0.f, 0.f, 0.f, 0.f, 0.f, 0.f};
  v8f acc[2][2] = {{zero, zero}, {zero, zero}};

  for (int tap = 0; tap < 9; ++tap) {
    const int dy = tap / 3 - 1, dx = tap % 3 - 1;
    const int yy = y + dy;
    const bool rowok = (yy >= 0) && (yy < H_);
    const f16* wt = w16 + tap * 16384;
    for (int c0 = 0; c0 < Cin; c0 += 32) {
      // A tile (contiguous in ic thanks to tap-major relayout)
      {
        int m  = tid >> 3;
        int kg = (tid & 7) << 2;
        f16x4 wv = *(const f16x4*)&wt[(size_t)(oc0 + m) * Cin + c0 + kg];
        int mt = m >> 4, mm = m & 15;
#pragma unroll
        for (int u = 0; u < 4; ++u) {
          int h, e; k_to_he(kg + u, h, e);
          Af[mt][(((h << 4) | mm) << 4) + e] = wv[u];
        }
      }
      // B tile: 32 x 256 with +/-1 column halo.  Load phase then scatter phase.
      f16 vals[32];
#pragma unroll
      for (int it = 0; it < 32; ++it) {
        int i = tid + (it << 8);
        int r = i >> 8, n = i & 255;
        int c  = c0 + r;
        int xx = n + dx;
        f16 vv = (f16)0.f;
        if (rowok && xx >= 0 && xx < W_) {
          const f16* src = (c < 64) ? inA : inB;
          vv = src[((size_t)b * 64 + (c & 63)) * HW_ + (size_t)yy * W_ + xx];
        }
        vals[it] = vv;
      }
#pragma unroll
      for (int it = 0; it < 32; ++it) {
        int i = tid + (it << 8);
        int r = i >> 8, n = i & 255;
        int h, e; k_to_he(r, h, e);
        int l = (h << 4) | (n & 15);
        Bf[((((n >> 4) << 5) + l) << 4) + e] = vals[it];
      }
      __syncthreads();

      v16h a0 = *(const v16h*)(Af[0] + (lane << 4));
      v16h a1 = *(const v16h*)(Af[1] + (lane << 4));
#pragma unroll
      for (int t2 = 0; t2 < 2; ++t2) {
        int nt = (wave << 1) + t2;
        v16h bf = *(const v16h*)(Bf + (((nt << 5) + lane) << 4));
        acc[0][t2] = __builtin_amdgcn_wmma_f32_16x16x32_f16(false, a0, false, bf,
                                                            (short)0, acc[0][t2], false, false);
        acc[1][t2] = __builtin_amdgcn_wmma_f32_16x16x32_f16(false, a1, false, bf,
                                                            (short)0, acc[1][t2], false, false);
      }
      __syncthreads();
    }
  }

#pragma unroll
  for (int mt = 0; mt < 2; ++mt)
#pragma unroll
    for (int t2 = 0; t2 < 2; ++t2) {
      int nt = (wave << 1) + t2;
#pragma unroll
      for (int r = 0; r < 8; ++r) {
        int m   = (mt << 4) + r + (half << 3);
        float v = acc[mt][t2][r];
        if (scale) v *= scale[oc0 + m] * bn_mul;
        if (shift) v += shift[oc0 + m];
        if (act == 1) v = gelu_exact(v);
        out[((size_t)b * Cout + oc0 + m) * HW_ + p0 + (nt << 4) + row] = (f16)v;
      }
    }
}

// ---------------------------------------------------------------------------
// Elementwise / reduction helpers
// ---------------------------------------------------------------------------
__global__ void f32_to_f16_k(const float* __restrict__ in, f16* __restrict__ out, size_t n) {
  size_t i = (size_t)blockIdx.x * 256 + threadIdx.x;
  if (i < n) out[i] = (f16)in[i];
}

// xh[b][c][h] = mean over w.  grid = B*C*H blocks of 256.
__global__ void pool_w_mean(const float* __restrict__ x, float* __restrict__ xh) {
  int rowid = blockIdx.x;
  const float* src = x + (size_t)rowid * W_;
  __shared__ float red[256];
  red[threadIdx.x] = src[threadIdx.x];
  __syncthreads();
  for (int s = 128; s > 0; s >>= 1) {
    if ((int)threadIdx.x < s) red[threadIdx.x] += red[threadIdx.x + s];
    __syncthreads();
  }
  if (threadIdx.x == 0) xh[rowid] = red[0] * (1.0f / 256.0f);
}

// xw[b][c][w] = mean over h.  grid = B*C blocks of 256 (thread = w).
__global__ void pool_h_mean(const float* __restrict__ x, float* __restrict__ xw) {
  int bc = blockIdx.x;
  const float* src = x + (size_t)bc * HW_;
  float s = 0.f;
  for (int h = 0; h < H_; ++h) s += src[h * W_ + threadIdx.x];
  xw[(size_t)bc * W_ + threadIdx.x] = s * (1.0f / 256.0f);
}

// y8[b][j][s] = hswish(bn(conv1([xh;xw]))).  grid = (8, B), block = 512.
__global__ void ca_stage1(const float* __restrict__ xh, const float* __restrict__ xw,
                          const float* __restrict__ w1, const float* __restrict__ b1,
                          const float* __restrict__ g, const float* __restrict__ bb,
                          float bn_mul, float* __restrict__ y8) {
  int j = blockIdx.x, b = blockIdx.y, s = threadIdx.x;
  float acc = b1[j];
  for (int c = 0; c < 64; ++c) {
    float iv = (s < 256) ? xh[((size_t)b * 64 + c) * 256 + s]
                         : xw[((size_t)b * 64 + c) * 256 + (s - 256)];
    acc += w1[j * 64 + c] * iv;
  }
  acc = acc * (g[j] * bn_mul) + bb[j];
  float r = fminf(fmaxf(acc + 3.f, 0.f), 6.f);
  y8[((size_t)b * 8 + j) * 512 + s] = acc * r * (1.0f / 6.0f);
}

// a_h / a_w = sigmoid(convh/convw(y8)).  grid = (64, B), block = 256.
__global__ void ca_stage2(const float* __restrict__ y8,
                          const float* __restrict__ wh, const float* __restrict__ bh,
                          const float* __restrict__ ww, const float* __restrict__ bw,
                          float* __restrict__ a_h, float* __restrict__ a_w) {
  int c = blockIdx.x, b = blockIdx.y, s = threadIdx.x;
  float ah = bh[c], aw = bw[c];
#pragma unroll
  for (int j = 0; j < 8; ++j) {
    ah += wh[c * 8 + j] * y8[((size_t)b * 8 + j) * 512 + s];
    aw += ww[c * 8 + j] * y8[((size_t)b * 8 + j) * 512 + 256 + s];
  }
  a_h[((size_t)b * 64 + c) * 256 + s] = sigmoidf_(ah);
  a_w[((size_t)b * 64 + c) * 256 + s] = sigmoidf_(aw);
}

// ca_out = x * a_h * a_w.  grid = (HW/256, 64, B).
__global__ void ca_apply(const float* __restrict__ x, const float* __restrict__ a_h,
                         const float* __restrict__ a_w, f16* __restrict__ ca_out) {
  int b = blockIdx.z, c = blockIdx.y;
  int p = blockIdx.x * 256 + threadIdx.x;
  int y = p >> 8, xx = p & 255;
  size_t base = (size_t)b * 64 + c;
  float v = x[base * HW_ + p] * a_h[base * 256 + y] * a_w[base * 256 + xx];
  ca_out[base * HW_ + p] = (f16)v;
}

// Depthwise 3x3, 384 channels.  grid = (HW/256, 384, B).
__global__ void dwconv3x3(const f16* __restrict__ in, const float* __restrict__ w,
                          f16* __restrict__ out) {
  int b = blockIdx.z, ch = blockIdx.y;
  int p = blockIdx.x * 256 + threadIdx.x;
  int y = p >> 8, x = p & 255;
  const f16* src = in + ((size_t)b * 384 + ch) * HW_;
  const float* wk = w + ch * 9;
  float acc = 0.f;
#pragma unroll
  for (int t = 0; t < 9; ++t) {
    int yy = y + t / 3 - 1, xx = x + t % 3 - 1;
    if (yy >= 0 && yy < H_ && xx >= 0 && xx < W_)
      acc += wk[t] * (float)src[yy * W_ + xx];
  }
  out[((size_t)b * 384 + ch) * HW_ + p] = (f16)acc;
}

// irfft2(rfft2(q)*rfft2(k)) over 8x8 patches == 8x8 circular convolution.
// grid = (1024 patches, 128 ch, B), block = 64 (one thread per patch pixel).
__global__ void patch_corr(const f16* __restrict__ hid, f16* __restrict__ o) {
  int b = blockIdx.z, ch = blockIdx.y;
  int ph = blockIdx.x >> 5, pw = blockIdx.x & 31;
  int t = threadIdx.x;
  int i = t >> 3, j = t & 7;
  __shared__ float qs[64], ks[64];
  int y = ph * 8 + i, x = pw * 8 + j;
  qs[t] = (float)hid[((size_t)b * 384 + ch) * HW_ + y * W_ + x];
  ks[t] = (float)hid[((size_t)b * 384 + 128 + ch) * HW_ + y * W_ + x];
  __syncthreads();
  float acc = 0.f;
#pragma unroll
  for (int a = 0; a < 8; ++a)
#pragma unroll
    for (int c = 0; c < 8; ++c)
      acc += qs[a * 8 + c] * ks[((i - a) & 7) * 8 + ((j - c) & 7)];
  o[((size_t)b * 128 + ch) * HW_ + y * W_ + x] = (f16)acc;
}

// Channel LayerNorm over 128 channels, then t = v * on.  grid = (HW/256, B).
__global__ void ln_mul(const f16* __restrict__ o, const f16* __restrict__ hdw,
                       const float* __restrict__ lw, const float* __restrict__ lb,
                       f16* __restrict__ t) {
  int b = blockIdx.y;
  int p = blockIdx.x * 256 + threadIdx.x;
  const f16* op = o + (size_t)b * 128 * HW_ + p;
  const f16* vp = hdw + ((size_t)b * 384 + 256) * HW_ + p;
  float s = 0.f, s2 = 0.f;
  for (int c = 0; c < 128; ++c) {
    float v = (float)op[(size_t)c * HW_];
    s += v; s2 += v * v;
  }
  float mu  = s * (1.0f / 128.0f);
  float var = s2 * (1.0f / 128.0f) - mu * mu;
  float rstd = rsqrtf(var + 1e-5f);
  for (int c = 0; c < 128; ++c) {
    float v  = (float)op[(size_t)c * HW_];
    float on = (v - mu) * rstd * lw[c] + lb[c];
    t[((size_t)b * 128 + c) * HW_ + p] = (f16)((float)vp[(size_t)c * HW_] * on);
  }
}

// z = x + rs * f2.  grid over n.
__global__ void residual_k(const float* __restrict__ x, const f16* __restrict__ f2,
                           const float* __restrict__ rs, float* __restrict__ z, size_t n) {
  size_t i = (size_t)blockIdx.x * 256 + threadIdx.x;
  if (i < n) z[i] = x[i] + rs[0] * (float)f2[i];
}

// CBAM global avg+max pooling per (b,c).  grid = B*C blocks of 256.
__global__ void cbam_pool(const float* __restrict__ z, float* __restrict__ avg,
                          float* __restrict__ mx) {
  int bc = blockIdx.x;
  const float* src = z + (size_t)bc * HW_;
  float s = 0.f, m = -3.4e38f;
  for (int i = threadIdx.x; i < HW_; i += 256) {
    float v = src[i]; s += v; m = fmaxf(m, v);
  }
  __shared__ float rs[256], rm[256];
  rs[threadIdx.x] = s; rm[threadIdx.x] = m;
  __syncthreads();
  for (int st = 128; st > 0; st >>= 1) {
    if ((int)threadIdx.x < st) {
      rs[threadIdx.x] += rs[threadIdx.x + st];
      rm[threadIdx.x]  = fmaxf(rm[threadIdx.x], rm[threadIdx.x + st]);
    }
    __syncthreads();
  }
  if (threadIdx.x == 0) { avg[bc] = rs[0] * (1.0f / 65536.0f); mx[bc] = rm[0]; }
}

// att = sigmoid(mlp(avg) + mlp(mx)).  grid = B, block = 64.
__global__ void cbam_att(const float* __restrict__ avg, const float* __restrict__ mx,
                         const float* __restrict__ m1, const float* __restrict__ m2,
                         float* __restrict__ att) {
  int b = blockIdx.x, c = threadIdx.x;
  __shared__ float h1[4];
  if (c < 4) {
    float sa = 0.f, sm = 0.f;
    for (int cc = 0; cc < 64; ++cc) {
      sa += m1[c * 64 + cc] * avg[b * 64 + cc];
      sm += m1[c * 64 + cc] * mx[b * 64 + cc];
    }
    h1[c] = fmaxf(sa, 0.f) + fmaxf(sm, 0.f);
  }
  __syncthreads();
  float o = 0.f;
#pragma unroll
  for (int r = 0; r < 4; ++r) o += m2[c * 4 + r] * h1[r];
  att[b * 64 + c] = sigmoidf_(o);
}

// stats = [mean_c, max_c] of z*att.  grid = (HW/256, B).
__global__ void sa_stats(const float* __restrict__ z, const float* __restrict__ att,
                         float* __restrict__ stats) {
  int b = blockIdx.y;
  int p = blockIdx.x * 256 + threadIdx.x;
  float s = 0.f, m = -3.4e38f;
  for (int c = 0; c < 64; ++c) {
    float v = z[((size_t)b * 64 + c) * HW_ + p] * att[b * 64 + c];
    s += v; m = fmaxf(m, v);
  }
  stats[((size_t)b * 2 + 0) * HW_ + p] = s * (1.0f / 64.0f);
  stats[((size_t)b * 2 + 1) * HW_ + p] = m;
}

// 7x7, pad 3, 2ch -> 1ch spatial-attention logits.  grid = (HW/256, B).
__global__ void sa_conv(const float* __restrict__ stats, const float* __restrict__ w,
                        float* __restrict__ logit) {
  int b = blockIdx.y;
  int p = blockIdx.x * 256 + threadIdx.x;
  int y = p >> 8, x = p & 255;
  float acc = 0.f;
  for (int ch = 0; ch < 2; ++ch)
    for (int ky = 0; ky < 7; ++ky) {
      int yy = y + ky - 3; if (yy < 0 || yy >= H_) continue;
      for (int kx = 0; kx < 7; ++kx) {
        int xx = x + kx - 3; if (xx < 0 || xx >= W_) continue;
        acc += w[(ch * 7 + ky) * 7 + kx] * stats[((size_t)b * 2 + ch) * HW_ + yy * W_ + xx];
      }
    }
  logit[(size_t)b * HW_ + p] = acc;
}

// out = gelu(z * att * sigmoid(logit)).  grid = (HW/256, 64, B).
__global__ void final_out(const float* __restrict__ z, const float* __restrict__ att,
                          const float* __restrict__ logit, float* __restrict__ out) {
  int b = blockIdx.z, c = blockIdx.y;
  int p = blockIdx.x * 256 + threadIdx.x;
  float v = z[((size_t)b * 64 + c) * HW_ + p] * att[b * 64 + c]
            * sigmoidf_(logit[(size_t)b * HW_ + p]);
  out[((size_t)b * 64 + c) * HW_ + p] = gelu_exact(v);
}

// ---------------------------------------------------------------------------
// Orchestration
// ---------------------------------------------------------------------------
extern "C" void kernel_launch(void* const* d_in, const int* in_sizes, int n_in,
                              void* d_out, int out_size, void* d_ws, size_t ws_size,
                              hipStream_t stream) {
  (void)in_sizes; (void)n_in; (void)out_size; (void)ws_size;
  const float* x        = (const float*)d_in[0];
  const float* ca1_w    = (const float*)d_in[1];
  const float* ca1_b    = (const float*)d_in[2];
  const float* ca_bn_g  = (const float*)d_in[3];
  const float* ca_bn_b  = (const float*)d_in[4];
  const float* cah_w    = (const float*)d_in[5];
  const float* cah_b    = (const float*)d_in[6];
  const float* caw_w    = (const float*)d_in[7];
  const float* caw_b    = (const float*)d_in[8];
  const float* hid_w    = (const float*)d_in[9];
  const float* dw_w     = (const float*)d_in[10];
  const float* proj_w   = (const float*)d_in[11];
  const float* ln_w     = (const float*)d_in[12];
  const float* ln_b     = (const float*)d_in[13];
  const float* fc1_w    = (const float*)d_in[14];
  const float* bn1_g    = (const float*)d_in[15];
  const float* bn1_b    = (const float*)d_in[16];
  const float* fc2_w    = (const float*)d_in[17];
  const float* bn2_g    = (const float*)d_in[18];
  const float* bn2_b    = (const float*)d_in[19];
  const float* mlp1_w   = (const float*)d_in[20];
  const float* mlp2_w   = (const float*)d_in[21];
  const float* sa_w     = (const float*)d_in[22];
  const float* res_s    = (const float*)d_in[23];
  float* out = (float*)d_out;

  const float bn_mul = 0.99999500003749969f;  // 1/sqrt(1 + 1e-5)

  // Workspace regions (lifetimes reused; see comments).
  char* ws = (char*)d_ws;
  const size_t R1_OFF = 0;                         // 201,326,592 B
  const size_t R2_OFF = 201326592;                 // 201,326,592 B
  const size_t R3_OFF = 402653184;                 //  67,108,864 B
  const size_t S_OFF  = 469762048;                 //   ~4.7 MB small buffers

  f16*  R1  = (f16*)(ws + R1_OFF);
  f16*  R2  = (f16*)(ws + R2_OFF);
  char* R3c = ws + R3_OFF;

  f16*   x16  = (f16*)R3c;                 // dead after hidden conv
  f16*   hid  = R1;                        // dead after depthwise
  f16*   hdw  = R2;                        // q/k/v; dead after ln_mul
  f16*   ocor = (f16*)R3c;                 // overwrites x16; dead after ln_mul
  f16*   t    = R1;                        // overwrites hid; dead after proj
  f16*   fsas = R2;                        // overwrites q; 33.5 MB
  f16*   cao  = R2 + 16777216;             // next 33.5 MB
  f16*   f1   = R1;                        // overwrites t; 67 MB
  f16*   f2   = R2 + 33554432;             // 33.5 MB
  float* z    = (float*)R3c;               // overwrites ocor; 67 MB

  float* xh    = (float*)(ws + S_OFF);     // B*C*H
  float* xw    = xh + 65536;               // B*C*W
  float* y8    = xw + 65536;               // B*8*512
  float* a_h   = y8 + 16384;
  float* a_w   = a_h + 65536;
  float* avgp  = a_w + 65536;              // B*C
  float* maxp  = avgp + 256;
  float* att   = maxp + 256;
  float* sst   = att + 256;                // B*2*HW
  float* salog = sst + 524288;             // B*HW
  // f16 weight caches (prep kernels fill these once per launch)
  f16* wh16  = (f16*)(salog + 262144);     // 384*64   = 24576
  f16* wp16  = wh16 + 24576;               // 64*128   = 8192
  f16* wf2   = wp16 + 8192;                // 64*128   = 8192
  f16* wf1   = wf2 + 8192;                 // 9*128*128 = 147456 (tap-major)

  const size_t nElem = (size_t)B_ * C_ * HW_;   // 16,777,216

  // 0) weight prep (tiny)
  prep_w_f16<<<dim3(96), dim3(256), 0, stream>>>(hid_w, wh16, 24576);
  prep_w_f16<<<dim3(32), dim3(256), 0, stream>>>(proj_w, wp16, 8192);
  prep_w_f16<<<dim3(32), dim3(256), 0, stream>>>(fc2_w, wf2, 8192);
  prep_w3x3<<<dim3(576), dim3(256), 0, stream>>>(fc1_w, wf1);
  // 1) x -> f16
  f32_to_f16_k<<<dim3(65536), dim3(256), 0, stream>>>(x, x16, nElem);
  // 2) coordinate-attention pools + tiny MLPs
  pool_w_mean<<<dim3(B_ * C_ * H_), dim3(256), 0, stream>>>(x, xh);
  pool_h_mean<<<dim3(B_ * C_), dim3(256), 0, stream>>>(x, xw);
  ca_stage1<<<dim3(8, B_), dim3(512), 0, stream>>>(xh, xw, ca1_w, ca1_b,
                                                   ca_bn_g, ca_bn_b, bn_mul, y8);
  ca_stage2<<<dim3(64, B_), dim3(256), 0, stream>>>(y8, cah_w, cah_b, caw_w, caw_b,
                                                    a_h, a_w);
  // 3) fsas_hidden: 1x1 conv 64 -> 384 (WMMA)
  conv1x1_wmma<<<dim3(HW_ / 256, 384 / 32, B_), dim3(256), 0, stream>>>(
      x16, wh16, hid, nullptr, nullptr, 1.f, 64, 384, 0);
  // 4) depthwise 3x3 on 384 channels
  dwconv3x3<<<dim3(HW_ / 256, 384, B_), dim3(256), 0, stream>>>(hid, dw_w, hdw);
  // 5) patch-wise FFT correlation == 8x8 circular convolution of q,k
  patch_corr<<<dim3(1024, 128, B_), dim3(64), 0, stream>>>(hdw, ocor);
  // 6) channel LayerNorm + multiply by v
  ln_mul<<<dim3(HW_ / 256, B_), dim3(256), 0, stream>>>(ocor, hdw, ln_w, ln_b, t);
  // 7) fsas_proj: 1x1 conv 128 -> 64 (WMMA)
  conv1x1_wmma<<<dim3(HW_ / 256, 64 / 32, B_), dim3(256), 0, stream>>>(
      t, wp16, fsas, nullptr, nullptr, 1.f, 128, 64, 0);
  // 8) ca_out = x * a_h * a_w  (R2 now free for it)
  ca_apply<<<dim3(HW_ / 256, 64, B_), dim3(256), 0, stream>>>(x, a_h, a_w, cao);
  // 9) fusion conv1: 3x3, virtual concat [ca_out ; fsas_out], + bn1 + exact GELU
  conv3x3_wmma<<<dim3(H_, 128 / 32, B_), dim3(256), 0, stream>>>(
      cao, fsas, wf1, f1, bn1_g, bn1_b, bn_mul, 1);
  // 10) fusion conv2: 1x1 128 -> 64 + bn2
  conv1x1_wmma<<<dim3(HW_ / 256, 64 / 32, B_), dim3(256), 0, stream>>>(
      f1, wf2, f2, bn2_g, bn2_b, bn_mul, 128, 64, 0);
  // 11) residual
  residual_k<<<dim3(65536), dim3(256), 0, stream>>>(x, f2, res_s, z, nElem);
  // 12) CBAM channel attention
  cbam_pool<<<dim3(B_ * C_), dim3(256), 0, stream>>>(z, avgp, maxp);
  cbam_att<<<dim3(B_), dim3(64), 0, stream>>>(avgp, maxp, mlp1_w, mlp2_w, att);
  // 13) CBAM spatial attention
  sa_stats<<<dim3(HW_ / 256, B_), dim3(256), 0, stream>>>(z, att, sst);
  sa_conv<<<dim3(HW_ / 256, B_), dim3(256), 0, stream>>>(sst, sa_w, salog);
  // 14) final: gelu(z * att * sigmoid(sa))
  final_out<<<dim3(HW_ / 256, 64, B_), dim3(256), 0, stream>>>(z, att, salog, out);
}
